// BiLSTMEnhanced_4020089389719
// MI455X (gfx1250) — matrix operand order repaired and checked by reading, not compiled, not run on hardware
//
#include <hip/hip_runtime.h>
#include <hip/hip_bf16.h>

typedef __attribute__((ext_vector_type(16))) __bf16        v16bf;
typedef __attribute__((ext_vector_type(8)))  float         v8f;
typedef __attribute__((ext_vector_type(4)))  unsigned int  u32x4;

#define B_   128
#define T_   128
#define E_   300
#define H_   256
#define L_   9
#define M_   16
#define CE_  30
#define NF_  50
#define K0PAD 416   // 400 padded to multiple of 32
#define K1    512
#define NG   1024   // 4*H

__device__ __forceinline__ unsigned short f2bf(float f) {
  union { float f; unsigned int u; } x; x.f = f;
  unsigned int u = x.u;
  unsigned int r = u + 0x7FFFu + ((u >> 16) & 1u);
  return (unsigned short)(r >> 16);
}
__device__ __forceinline__ float bf2f(unsigned short h) {
  union { unsigned int u; float f; } x; x.u = ((unsigned int)h) << 16;
  return x.f;
}
__device__ __forceinline__ float sigm(float x) { return 1.0f / (1.0f + __expf(-x)); }

// ---------------------------------------------------------------------------
// Word embedding gather + char CNN -> x_bf16 [B*T, 416] (cols 400..415 zero)
// ---------------------------------------------------------------------------
__global__ void featurize(const int* __restrict__ word_ids, const int* __restrict__ char_ids,
                          const float* __restrict__ word_emb, const float* __restrict__ char_emb,
                          const float* __restrict__ c2w, const float* __restrict__ c2b,
                          const float* __restrict__ c3w, const float* __restrict__ c3b,
                          unsigned short* __restrict__ xbf) {
  __shared__ float ch[M_][CE_];
  int bt = blockIdx.x;
  int tid = threadIdx.x;
  for (int i = tid; i < M_ * CE_; i += blockDim.x) {
    int m = i / CE_, d = i % CE_;
    int cid = char_ids[bt * M_ + m];
    ch[m][d] = char_emb[cid * CE_ + d];
  }
  __syncthreads();
  int wid = word_ids[bt];
  unsigned short* xr = xbf + (size_t)bt * K0PAD;
  for (int j = tid; j < E_; j += blockDim.x)
    xr[j] = f2bf(word_emb[(size_t)wid * E_ + j]);
  if (tid < NF_) {                       // conv k=2, relu, max-pool
    int f = tid;
    float best = -3.0e38f;
    for (int p = 0; p < M_ - 1; ++p) {
      float y = c2b[f];
      for (int d = 0; d < CE_; ++d)
        y += ch[p][d]   * c2w[f * CE_ * 2 + d * 2 + 0]
           + ch[p+1][d] * c2w[f * CE_ * 2 + d * 2 + 1];
      best = fmaxf(best, y);
    }
    xr[E_ + f] = f2bf(fmaxf(best, 0.0f));
  } else if (tid < 2 * NF_) {            // conv k=3
    int f = tid - NF_;
    float best = -3.0e38f;
    for (int p = 0; p < M_ - 2; ++p) {
      float y = c3b[f];
      for (int d = 0; d < CE_; ++d)
        y += ch[p][d]   * c3w[f * CE_ * 3 + d * 3 + 0]
           + ch[p+1][d] * c3w[f * CE_ * 3 + d * 3 + 1]
           + ch[p+2][d] * c3w[f * CE_ * 3 + d * 3 + 2];
      best = fmaxf(best, y);
    }
    xr[E_ + NF_ + f] = f2bf(fmaxf(best, 0.0f));
  }
  if (tid < K0PAD - (E_ + 2 * NF_))      // zero pad cols 400..415
    xr[E_ + 2 * NF_ + tid] = 0;
}

// fp32 -> bf16 weight convert, with zero padding of K dimension
__global__ void cvt_pad(const float* __restrict__ src, unsigned short* __restrict__ dst,
                        int K, int Kpad, int total) {
  int i = blockIdx.x * blockDim.x + threadIdx.x;
  if (i >= total) return;
  int n = i / Kpad, k = i % Kpad;
  dst[i] = (k < K) ? f2bf(src[(size_t)n * K + k]) : (unsigned short)0;
}

__global__ void addvec(const float* __restrict__ a, const float* __restrict__ b,
                       float* __restrict__ o, int n) {
  int i = blockIdx.x * blockDim.x + threadIdx.x;
  if (i < n) o[i] = a[i] + b[i];
}

__global__ void zero_u32(unsigned int* __restrict__ p, int n) {
  int i = blockIdx.x * blockDim.x + threadIdx.x;
  if (i < n) p[i] = 0u;
}

// ---------------------------------------------------------------------------
// Batched input projection, 2x2 register-blocked WMMA:
//   Y[bt,n] = bf16( sum_k X[bt,k]*W[n,k] + bias[n] )
// One wave computes a 32x32 output block (4 accumulators, 4 WMMAs / k-step).
// ---------------------------------------------------------------------------
__global__ void gemm_inproj(const unsigned short* __restrict__ X,   // [Mrows, K] bf16
                            const unsigned short* __restrict__ W,   // [NG, K]    bf16
                            const float* __restrict__ bias,         // [NG]
                            unsigned short* __restrict__ Y,         // [Mrows, NG] bf16
                            int Mrows, int K) {
  int wid  = blockIdx.x * (blockDim.x >> 5) + (threadIdx.x >> 5);
  int lane = threadIdx.x & 31;
  int half = lane >> 4, l15 = lane & 15;
  int mpairs = Mrows >> 5;                 // # of 32-row blocks
  int mp = wid % mpairs, np = wid / mpairs;
  int m0 = mp << 5, n0 = np << 5;
  const unsigned short* pa0 = X + (size_t)(m0 + l15) * K;
  const unsigned short* pa1 = pa0 + (size_t)16 * K;
  const unsigned short* pb0 = W + (size_t)(n0 + l15) * K;
  const unsigned short* pb1 = pb0 + (size_t)16 * K;
  union Frag { u32x4 u[2]; v16bf v; };
  v8f acc00 = {}, acc01 = {}, acc10 = {}, acc11 = {};
  for (int k0 = 0; k0 < K; k0 += 32) {
    Frag fa0, fa1, fb0, fb1;
    int ao = k0 + half * 8;
    int bo = k0 + half * 8;
    // A 16x32 bf16: lane=row; K = {half*8+0..7} U {16+half*8+0..7}
    fa0.u[0] = *reinterpret_cast<const u32x4*>(pa0 + ao);
    fa0.u[1] = *reinterpret_cast<const u32x4*>(pa0 + ao + 16);
    fa1.u[0] = *reinterpret_cast<const u32x4*>(pa1 + ao);
    fa1.u[1] = *reinterpret_cast<const u32x4*>(pa1 + ao + 16);
    // B 32x16 bf16: lane=col; K = {half*8+0..7} U {16+half*8+0..7}
    fb0.u[0] = *reinterpret_cast<const u32x4*>(pb0 + bo);
    fb0.u[1] = *reinterpret_cast<const u32x4*>(pb0 + bo + 16);
    fb1.u[0] = *reinterpret_cast<const u32x4*>(pb1 + bo);
    fb1.u[1] = *reinterpret_cast<const u32x4*>(pb1 + bo + 16);
    acc00 = __builtin_amdgcn_wmma_f32_16x16x32_bf16(false, fa0.v, false, fb0.v, (short)0, acc00, false, false);
    acc01 = __builtin_amdgcn_wmma_f32_16x16x32_bf16(false, fa0.v, false, fb1.v, (short)0, acc01, false, false);
    acc10 = __builtin_amdgcn_wmma_f32_16x16x32_bf16(false, fa1.v, false, fb0.v, (short)0, acc10, false, false);
    acc11 = __builtin_amdgcn_wmma_f32_16x16x32_bf16(false, fa1.v, false, fb1.v, (short)0, acc11, false, false);
  }
  int col0 = n0 + l15, col1 = col0 + 16;
  float bv0 = bias[col0], bv1 = bias[col1];
  for (int r = 0; r < 8; ++r) {
    int row0 = m0 + half * 8 + r;           // C layout: M = r + half*8
    int row1 = row0 + 16;
    Y[(size_t)row0 * NG + col0] = f2bf(acc00[r] + bv0);
    Y[(size_t)row0 * NG + col1] = f2bf(acc01[r] + bv1);
    Y[(size_t)row1 * NG + col0] = f2bf(acc10[r] + bv0);
    Y[(size_t)row1 * NG + col1] = f2bf(acc11[r] + bv1);
  }
}

// ---------------------------------------------------------------------------
// Recurrent step, 2x2 register-blocked:
//   g[dir,b,n] = xs[dir][b*T+t, n] + sum_k h[dir,b,k]*Whh[dir,n,k]
// 2 dirs x (4 x 32) 32x32-blocks = 256 waves. K = 256.
// ---------------------------------------------------------------------------
__global__ void lstm_step_gemm(const unsigned short* __restrict__ xsF,
                               const unsigned short* __restrict__ xsB,
                               const unsigned short* __restrict__ hstate, // [2][B_][H_] bf16
                               const unsigned short* __restrict__ Whh,    // [2][NG][H_] bf16
                               float* __restrict__ g,                     // [2][B_][NG]
                               int tf, int tb) {
  int wid  = blockIdx.x * (blockDim.x >> 5) + (threadIdx.x >> 5);
  int lane = threadIdx.x & 31;
  int half = lane >> 4, l15 = lane & 15;
  int dir  = wid >> 7;                     // 128 32x32-blocks per dir
  int tile = wid & 127;
  int mp = tile & 3, np = tile >> 2;
  int m0 = mp << 5, n0 = np << 5;
  const unsigned short* pa0 = hstate + (size_t)dir * B_ * H_ + (size_t)(m0 + l15) * H_;
  const unsigned short* pa1 = pa0 + (size_t)16 * H_;
  const unsigned short* pb0 = Whh + (size_t)dir * NG * H_ + (size_t)(n0 + l15) * H_;
  const unsigned short* pb1 = pb0 + (size_t)16 * H_;
  union Frag { u32x4 u[2]; v16bf v; };
  v8f acc00 = {}, acc01 = {}, acc10 = {}, acc11 = {};
  for (int k0 = 0; k0 < H_; k0 += 32) {
    Frag fa0, fa1, fb0, fb1;
    int ao = k0 + half * 8;
    int bo = k0 + half * 8;
    fa0.u[0] = *reinterpret_cast<const u32x4*>(pa0 + ao);
    fa0.u[1] = *reinterpret_cast<const u32x4*>(pa0 + ao + 16);
    fa1.u[0] = *reinterpret_cast<const u32x4*>(pa1 + ao);
    fa1.u[1] = *reinterpret_cast<const u32x4*>(pa1 + ao + 16);
    fb0.u[0] = *reinterpret_cast<const u32x4*>(pb0 + bo);
    fb0.u[1] = *reinterpret_cast<const u32x4*>(pb0 + bo + 16);
    fb1.u[0] = *reinterpret_cast<const u32x4*>(pb1 + bo);
    fb1.u[1] = *reinterpret_cast<const u32x4*>(pb1 + bo + 16);
    acc00 = __builtin_amdgcn_wmma_f32_16x16x32_bf16(false, fa0.v, false, fb0.v, (short)0, acc00, false, false);
    acc01 = __builtin_amdgcn_wmma_f32_16x16x32_bf16(false, fa0.v, false, fb1.v, (short)0, acc01, false, false);
    acc10 = __builtin_amdgcn_wmma_f32_16x16x32_bf16(false, fa1.v, false, fb0.v, (short)0, acc10, false, false);
    acc11 = __builtin_amdgcn_wmma_f32_16x16x32_bf16(false, fa1.v, false, fb1.v, (short)0, acc11, false, false);
  }
  int col0 = n0 + l15, col1 = col0 + 16;
  int tIdx = dir ? tb : tf;
  const unsigned short* xs = dir ? xsB : xsF;
  float* gd = g + (size_t)dir * B_ * NG;
  for (int r = 0; r < 8; ++r) {
    int b0 = m0 + half * 8 + r;
    int b1 = b0 + 16;
    size_t x0 = ((size_t)b0 * T_ + tIdx) * NG;
    size_t x1 = ((size_t)b1 * T_ + tIdx) * NG;
    gd[(size_t)b0 * NG + col0] = acc00[r] + bf2f(xs[x0 + col0]);
    gd[(size_t)b0 * NG + col1] = acc01[r] + bf2f(xs[x0 + col1]);
    gd[(size_t)b1 * NG + col0] = acc10[r] + bf2f(xs[x1 + col0]);
    gd[(size_t)b1 * NG + col1] = acc11[r] + bf2f(xs[x1 + col1]);
  }
}

// gate nonlinearity: i,f,g,o -> c,h ; write h to state (bf16) and sequence out
__global__ void lstm_step_gates(const float* __restrict__ g, float* __restrict__ cst,
                                unsigned short* __restrict__ hst,
                                unsigned short* __restrict__ hout,   // [B*T, 2H] bf16
                                int tf, int tb) {
  int idx = blockIdx.x * blockDim.x + threadIdx.x;   // 2*B_*H_ = 65536
  int dir = idx >> 15;
  int rem = idx & 32767;
  int b = rem >> 8, j = rem & 255;
  const float* gr = g + (size_t)dir * B_ * NG + (size_t)b * NG;
  float i_ = gr[j], f_ = gr[H_ + j], gg = gr[2 * H_ + j], o_ = gr[3 * H_ + j];
  size_t ci = (size_t)dir * B_ * H_ + (size_t)b * H_ + j;
  float c = sigm(f_) * cst[ci] + sigm(i_) * tanhf(gg);
  float h = sigm(o_) * tanhf(c);
  cst[ci] = c;
  unsigned short hb = f2bf(h);
  hst[ci] = hb;
  int tIdx = dir ? tb : tf;
  hout[((size_t)b * T_ + tIdx) * (2 * H_) + dir * H_ + j] = hb;
}

// emissions [B*T, L] = h1 @ fc_w^T + fc_b  (N=9, plain dot)
__global__ void fc_emissions(const unsigned short* __restrict__ h1,
                             const float* __restrict__ fw, const float* __restrict__ fb,
                             float* __restrict__ em) {
  int idx = blockIdx.x * blockDim.x + threadIdx.x;
  if (idx >= B_ * T_ * L_) return;
  int bt = idx / L_, l = idx % L_;
  const unsigned short* hr = h1 + (size_t)bt * (2 * H_);
  const float* wr = fw + (size_t)l * (2 * H_);
  float s = fb[l];
  for (int k = 0; k < 2 * H_; ++k) s += bf2f(hr[k]) * wr[k];
  em[idx] = s;
}

// batched Viterbi decode, one thread per batch element (L=9)
__global__ void viterbi(const float* __restrict__ em, const float* __restrict__ trans,
                        const float* __restrict__ st, const float* __restrict__ en,
                        int* __restrict__ bp, float* __restrict__ out) {
  __shared__ float tr[L_ * L_];
  int tid = threadIdx.x;
  if (tid < L_ * L_) tr[tid] = trans[tid];
  __syncthreads();
  int b = tid;
  if (b >= B_) return;
  float v[L_];
  for (int l = 0; l < L_; ++l) v[l] = st[l] + em[((size_t)b * T_) * L_ + l];
  for (int t = 1; t < T_; ++t) {
    float nv[L_];
    for (int l2 = 0; l2 < L_; ++l2) {
      float best = -3.0e38f; int arg = 0;
      for (int l1 = 0; l1 < L_; ++l1) {
        float s = v[l1] + tr[l1 * L_ + l2];
        if (s > best) { best = s; arg = l1; }
      }
      bp[((size_t)(t - 1) * B_ + b) * L_ + l2] = arg;
      nv[l2] = best + em[((size_t)b * T_ + t) * L_ + l2];
    }
    for (int l = 0; l < L_; ++l) v[l] = nv[l];
  }
  int last = 0; float best = -3.0e38f;
  for (int l = 0; l < L_; ++l) { float s = v[l] + en[l]; if (s > best) { best = s; last = l; } }
  out[(size_t)b * T_ + T_ - 1] = (float)last;
  int cur = last;
  for (int t = T_ - 2; t >= 0; --t) {
    cur = bp[((size_t)t * B_ + b) * L_ + cur];
    out[(size_t)b * T_ + t] = (float)cur;
  }
}

extern "C" void kernel_launch(void* const* d_in, const int* in_sizes, int n_in,
                              void* d_out, int out_size, void* d_ws, size_t ws_size,
                              hipStream_t stream) {
  (void)in_sizes; (void)n_in; (void)out_size; (void)ws_size;
  const int*   word_ids = (const int*)d_in[0];
  const int*   char_ids = (const int*)d_in[1];
  const float* word_emb = (const float*)d_in[3];
  const float* char_emb = (const float*)d_in[4];
  const float* c2w = (const float*)d_in[5];
  const float* c2b = (const float*)d_in[6];
  const float* c3w = (const float*)d_in[7];
  const float* c3b = (const float*)d_in[8];
  const float* w_ih[4] = {(const float*)d_in[9],  (const float*)d_in[13],
                          (const float*)d_in[17], (const float*)d_in[21]};
  const float* w_hh[4] = {(const float*)d_in[10], (const float*)d_in[14],
                          (const float*)d_in[18], (const float*)d_in[22]};
  const float* b_ih[4] = {(const float*)d_in[11], (const float*)d_in[15],
                          (const float*)d_in[19], (const float*)d_in[23]};
  const float* b_hh[4] = {(const float*)d_in[12], (const float*)d_in[16],
                          (const float*)d_in[20], (const float*)d_in[24]};
  const float* fc_w   = (const float*)d_in[25];
  const float* fc_b   = (const float*)d_in[26];
  const float* trans  = (const float*)d_in[27];
  const float* startt = (const float*)d_in[28];
  const float* endt   = (const float*)d_in[29];

  char* ws = (char*)d_ws;
  size_t off = 0;
  auto alloc = [&](size_t bytes) -> char* {
    off = (off + 255) & ~(size_t)255;
    char* p = ws + off; off += bytes; return p;
  };
  const size_t BT = (size_t)B_ * T_;
  unsigned short* xbf  = (unsigned short*)alloc(BT * K0PAD * 2);
  unsigned short* h0bf = (unsigned short*)alloc(BT * 2 * H_ * 2);
  unsigned short* h1bf = (unsigned short*)alloc(BT * 2 * H_ * 2);
  unsigned short* xsF  = (unsigned short*)alloc(BT * NG * 2);
  unsigned short* xsB  = (unsigned short*)alloc(BT * NG * 2);
  unsigned short* Wih0 = (unsigned short*)alloc((size_t)2 * NG * K0PAD * 2);
  unsigned short* Wih1 = (unsigned short*)alloc((size_t)2 * NG * K1 * 2);
  unsigned short* Whh  = (unsigned short*)alloc((size_t)4 * NG * H_ * 2);
  float*          biasC= (float*)alloc((size_t)4 * NG * 4);
  unsigned short* hst  = (unsigned short*)alloc((size_t)2 * B_ * H_ * 2);
  float*          cst  = (float*)alloc((size_t)2 * B_ * H_ * 4);
  float*          g    = (float*)alloc((size_t)2 * B_ * NG * 4);
  float*          em   = (float*)alloc(BT * L_ * 4);
  int*            bp   = (int*)alloc((size_t)(T_ - 1) * B_ * L_ * 4);

  // features: word emb + char CNN -> bf16 x [16384, 416]
  featurize<<<B_ * T_, 128, 0, stream>>>(word_ids, char_ids, word_emb, char_emb,
                                         c2w, c2b, c3w, c3b, xbf);

  // weights -> bf16 (with K padding for layer 0), combined biases
  {
    int tot0 = NG * K0PAD;
    cvt_pad<<<(tot0 + 255) / 256, 256, 0, stream>>>(w_ih[0], Wih0, 400, K0PAD, tot0);
    cvt_pad<<<(tot0 + 255) / 256, 256, 0, stream>>>(w_ih[1], Wih0 + (size_t)NG * K0PAD, 400, K0PAD, tot0);
    int tot1 = NG * K1;
    cvt_pad<<<(tot1 + 255) / 256, 256, 0, stream>>>(w_ih[2], Wih1, K1, K1, tot1);
    cvt_pad<<<(tot1 + 255) / 256, 256, 0, stream>>>(w_ih[3], Wih1 + (size_t)NG * K1, K1, K1, tot1);
    int toth = NG * H_;
    for (int d = 0; d < 4; ++d)
      cvt_pad<<<(toth + 255) / 256, 256, 0, stream>>>(w_hh[d], Whh + (size_t)d * NG * H_, H_, H_, toth);
    for (int d = 0; d < 4; ++d)
      addvec<<<(NG + 255) / 256, 256, 0, stream>>>(b_ih[d], b_hh[d], biasC + (size_t)d * NG, NG);
  }

  for (int l = 0; l < 2; ++l) {
    const unsigned short* X    = (l == 0) ? xbf : h0bf;
    int                   K    = (l == 0) ? K0PAD : K1;
    const unsigned short* Wih  = (l == 0) ? Wih0 : Wih1;
    unsigned short*       hout = (l == 0) ? h0bf : h1bf;
    const float*          bF   = biasC + (size_t)(2 * l) * NG;
    const float*          bB   = biasC + (size_t)(2 * l + 1) * NG;
    const unsigned short* WhhL = Whh + (size_t)(2 * l) * NG * H_;

    // batched input projections (2x2-blocked WMMA): 512 x 32 blocks, 4 waves/block
    gemm_inproj<<<4096, 128, 0, stream>>>(X, Wih,                  bF, xsF, B_ * T_, K);
    gemm_inproj<<<4096, 128, 0, stream>>>(X, Wih + (size_t)NG * K, bB, xsB, B_ * T_, K);

    // zero h (bf16) and c (f32) state
    zero_u32<<<((2 * B_ * H_ / 2) + 255) / 256, 256, 0, stream>>>((unsigned int*)hst, 2 * B_ * H_ / 2);
    zero_u32<<<((2 * B_ * H_) + 255) / 256, 256, 0, stream>>>((unsigned int*)cst, 2 * B_ * H_);

    // sequential recurrence, both directions per launch (WMMA on h @ Whh^T)
    for (int t = 0; t < T_; ++t) {
      lstm_step_gemm<<<64, 128, 0, stream>>>(xsF, xsB, hst, WhhL, g, t, T_ - 1 - t);
      lstm_step_gates<<<256, 256, 0, stream>>>(g, cst, hst, hout, t, T_ - 1 - t);
    }
  }

  int tote = B_ * T_ * L_;
  fc_emissions<<<(tote + 255) / 256, 256, 0, stream>>>(h1bf, fc_w, fc_b, em);
  viterbi<<<1, 128, 0, stream>>>(em, trans, startt, endt, bp, (float*)d_out);
}
